// FastAttention_16724602651112
// MI455X (gfx1250) — matrix operand order repaired and checked by reading, not compiled, not run on hardware
//
#include <hip/hip_runtime.h>
#include <hip/hip_bf16.h>

// ---------------- types ----------------
typedef __bf16 v16bf __attribute__((ext_vector_type(16)));
typedef __bf16 v8bf  __attribute__((ext_vector_type(8)));
typedef float  v8f   __attribute__((ext_vector_type(8)));

// ---------------- problem constants ----------------
#define N_ROWS 65536
#define DIM    64
#define RFEAT  256
#define DV     64
#define DVE    80                 // dv + ones-column tile (s) padded to 5 N-tiles
#define NB     128                // blocks for z-partial kernel
#define CHUNK  (N_ROWS / NB)      // 512 rows per block
#define DSCALE 0.35355339059327379f   // 64^-0.25
#define RATIO  0.0625f                // 256^-0.5

static __device__ __forceinline__ v8f wmma_bf16(v16bf a, v16bf b, v8f c) {
  return __builtin_amdgcn_wmma_f32_16x16x32_bf16(false, a, false, b, (short)0, c,
                                                 false, false);
}

// A-fragment (16x32 bf16) from row-major f32 data, scaled by d^-0.25,
// accumulating this lane's partial sum of squares of the scaled row.
// Lane l<16: row = base+l, K = kbase+{0..15}; lane>=16: K +16.
static __device__ __forceinline__ v16bf load_data_frag(const float* rowp, int kbase,
                                                       float& ss) {
  const int l = threadIdx.x & 31;
  const int k0 = kbase + ((l >> 4) << 4);
  v16bf a;
#pragma unroll
  for (int i = 0; i < 8; ++i) {
    float x0 = DSCALE * rowp[k0 + i];
    float x1 = DSCALE * rowp[k0 + 8 + i];
    ss += x0 * x0 + x1 * x1;
    a[i]     = (__bf16)x0;
    a[i + 8] = (__bf16)x1;
  }
  return a;
}

// B-fragment (32x16 bf16): lane holds one column (= one contiguous row of the
// row-major source), 16 contiguous K values starting at kbase + (lane/16)*16.
static __device__ __forceinline__ v16bf load_proj_frag(const float* colp, int kbase) {
  const int l = threadIdx.x & 31;
  const int k0 = kbase + ((l >> 4) << 4);
  v16bf b;
#pragma unroll
  for (int i = 0; i < 16; ++i) b[i] = (__bf16)colp[k0 + i];
  return b;
}

// A-fragment from a contiguous bf16 row (LDS or global), two b128 chunks.
static __device__ __forceinline__ v16bf load_bf16_a_frag(const __bf16* ap, int kbase) {
  const int l = threadIdx.x & 31;
  const int k0 = kbase + ((l >> 4) << 4);
  v8bf lo = *(const v8bf*)(ap + k0);
  v8bf hi = *(const v8bf*)(ap + k0 + 8);
  return __builtin_shufflevector(lo, hi, 0, 1, 2, 3, 4, 5, 6, 7, 8, 9, 10, 11, 12,
                                 13, 14, 15);
}

// B-fragment from a contiguous bf16 column-data row, 16 contiguous values.
static __device__ __forceinline__ v16bf load_bf16_b_frag(const __bf16* zp, int kbase) {
  const int l = threadIdx.x & 31;
  const int k0 = kbase + ((l >> 4) << 4);
  v8bf lo = *(const v8bf*)(zp + k0);
  v8bf hi = *(const v8bf*)(zp + k0 + 8);
  return __builtin_shufflevector(lo, hi, 0, 1, 2, 3, 4, 5, 6, 7, 8, 9, 10, 11, 12,
                                 13, 14, 15);
}

// =====================================================================
// Kernel 1: per-block partial  z^T_ext = [v|1]^T k_hat  over a 512-row chunk.
// K featurization fused on-chip: k_hat strip is staged transposed in LDS.
// =====================================================================
__global__ __launch_bounds__(256) void favor_kz_partial(
    const float* __restrict__ kmat, const float* __restrict__ vmat,
    const float* __restrict__ proj, float* __restrict__ part) {
  __shared__ __bf16 khT[RFEAT * 64];  // k_hat^T strip [r][64 rows], 32 KB

  const int w = threadIdx.x >> 5;
  const int l = threadIdx.x & 31;
  const int lo16 = l & 15;
  const int hi = l >> 4;
  const int blk = blockIdx.x;

  // ones-column B fragment for the extended v (dve tile 4: col 64 = 1, 65..79 = 0)
  v16bf bones;
  {
    __bf16 one = (__bf16)(lo16 == 0 ? 1.0f : 0.0f);
#pragma unroll
    for (int i = 0; i < 16; ++i) bones[i] = one;
  }

  v8f cz[2][5];
  {
    v8f zero = {};
#pragma unroll
    for (int rt = 0; rt < 2; ++rt)
#pragma unroll
      for (int nt = 0; nt < 5; ++nt) cz[rt][nt] = zero;
  }

  const int rowtile = w >> 1;        // wave pair shares a 16-row tile
  const int ct0 = (w & 1) * 8;       // even wave: r-tiles 0..7, odd: 8..15

  for (int sub = 0; sub < CHUNK / 64; ++sub) {
    // ---- Stage A: kernelize 64 K-rows -> LDS as k_hat^T (bf16)
    const int nbase = blk * CHUNK + sub * 64 + rowtile * 16;
    const float* rowp = kmat + (size_t)(nbase + lo16) * DIM;
    float ss = 0.0f;
    v16bf a0 = load_data_frag(rowp, 0, ss);
    v16bf a1 = load_data_frag(rowp, 32, ss);
    float h = __expf(-0.5f * (ss + __shfl_xor(ss, 16, 32)));
    float hv[8];
#pragma unroll
    for (int j = 0; j < 8; ++j) hv[j] = __shfl(h, j + (hi << 3), 32);

#pragma unroll
    for (int c = 0; c < 8; ++c) {
      const int ct = ct0 + c;
      const float* colp = proj + (size_t)(ct * 16 + lo16) * DIM;
      v16bf b0 = load_proj_frag(colp, 0);
      v16bf b1 = load_proj_frag(colp, 32);
      v8f acc = {};
      acc = wmma_bf16(a0, b0, acc);
      acc = wmma_bf16(a1, b1, acc);
      v8bf pk;
#pragma unroll
      for (int j = 0; j < 8; ++j)
        pk[j] = (__bf16)(RATIO * hv[j] * __expf(acc[j]));
      // C rows are contiguous in n -> single b128 LDS store per tile
      *(v8bf*)(khT + (ct * 16 + lo16) * 64 + rowtile * 16 + (hi << 3)) = pk;
    }
    __syncthreads();

    // ---- Stage B: accumulate z^T partial, wave w owns r rows [32w, 32w+32)
#pragma unroll
    for (int kc = 0; kc < 2; ++kc) {
      const int kb = kc * 32;
      v16bf a[2];
#pragma unroll
      for (int rt = 0; rt < 2; ++rt)
        a[rt] = load_bf16_a_frag(khT + ((w * 2 + rt) * 16 + lo16) * 64, kb);

      const int krow0 = blk * CHUNK + sub * 64 + kb + (hi << 4);
      v16bf b[5];
#pragma unroll
      for (int nt = 0; nt < 4; ++nt) {
#pragma unroll
        for (int i = 0; i < 16; ++i)
          b[nt][i] = (__bf16)vmat[(size_t)(krow0 + i) * DV + nt * 16 + lo16];
      }
      b[4] = bones;

#pragma unroll
      for (int rt = 0; rt < 2; ++rt)
#pragma unroll
        for (int nt = 0; nt < 5; ++nt) cz[rt][nt] = wmma_bf16(a[rt], b[nt], cz[rt][nt]);
    }
    __syncthreads();
  }

  // ---- store partials as [blk][dve][r] (lane-contiguous in r -> b128 stores)
  float* pb = part + (size_t)blk * (DVE * RFEAT);
#pragma unroll
  for (int rt = 0; rt < 2; ++rt) {
#pragma unroll
    for (int nt = 0; nt < 5; ++nt) {
      const int dve = nt * 16 + lo16;
      const int r0 = (w * 2 + rt) * 16 + (hi << 3);
      float* pp = pb + dve * RFEAT + r0;
#pragma unroll
      for (int j = 0; j < 8; ++j) pp[j] = cz[rt][nt][j];
    }
  }
}

// =====================================================================
// Kernel 2: deterministic reduction of NB partials -> z^T_ext bf16 [80][256]
// (row 64 carries s = column-sums of k_hat, f32-accumulated)
// =====================================================================
__global__ __launch_bounds__(256) void favor_z_reduce(const float* __restrict__ part,
                                                      __bf16* __restrict__ zTe) {
  const int dve = blockIdx.x;
  const int r = threadIdx.x;
  float s = 0.0f;
  for (int p = 0; p < NB; ++p)
    s += part[(size_t)p * (DVE * RFEAT) + dve * RFEAT + r];
  zTe[dve * RFEAT + r] = (__bf16)s;
}

// =====================================================================
// Kernel 3: fused  q -> q_hat (WMMA, LDS strip) -> W_ext = q_hat @ z_ext
// -> out = W[:, :64] / W[:, 64]
// =====================================================================
__global__ __launch_bounds__(256) void favor_qw_out(const float* __restrict__ qmat,
                                                    const float* __restrict__ proj,
                                                    const __bf16* __restrict__ zTe,
                                                    float* __restrict__ out) {
  __shared__ __bf16 qh[8 * 16 * RFEAT];  // per-wave [16][256] strips, 64 KB

  const int w = threadIdx.x >> 5;
  const int l = threadIdx.x & 31;
  const int lo16 = l & 15;
  const int hi = l >> 4;
  const int rowbase = blockIdx.x * 128 + w * 16;

  // ---- featurize 16 Q rows into LDS strip (bf16, row-major [16][256])
  const float* rowp = qmat + (size_t)(rowbase + lo16) * DIM;
  float ss = 0.0f;
  v16bf a0 = load_data_frag(rowp, 0, ss);
  v16bf a1 = load_data_frag(rowp, 32, ss);
  float h = __expf(-0.5f * (ss + __shfl_xor(ss, 16, 32)));
  float hv[8];
#pragma unroll
  for (int j = 0; j < 8; ++j) hv[j] = __shfl(h, j + (hi << 3), 32);

  __bf16* strip = qh + w * (16 * RFEAT);
#pragma unroll
  for (int ct = 0; ct < 16; ++ct) {
    const float* colp = proj + (size_t)(ct * 16 + lo16) * DIM;
    v16bf b0 = load_proj_frag(colp, 0);
    v16bf b1 = load_proj_frag(colp, 32);
    v8f acc = {};
    acc = wmma_bf16(a0, b0, acc);
    acc = wmma_bf16(a1, b1, acc);
#pragma unroll
    for (int j = 0; j < 8; ++j)
      strip[(j + (hi << 3)) * RFEAT + ct * 16 + lo16] =
          (__bf16)(RATIO * hv[j] * __expf(acc[j]));
  }
  __syncthreads();

  // ---- W_ext[16,80] = q_hat[16,256] @ z_ext[256,80]  (K = 256, 8 k-steps)
  v8f cw[5];
  {
    v8f zero = {};
#pragma unroll
    for (int nt = 0; nt < 5; ++nt) cw[nt] = zero;
  }
  const __bf16* arow = strip + lo16 * RFEAT;
#pragma unroll
  for (int kc = 0; kc < 8; ++kc) {
    const int kb = kc * 32;
    v16bf a = load_bf16_a_frag(arow, kb);
#pragma unroll
    for (int nt = 0; nt < 5; ++nt) {
      v16bf b = load_bf16_b_frag(zTe + (nt * 16 + lo16) * RFEAT, kb);
      cw[nt] = wmma_bf16(a, b, cw[nt]);
    }
  }

  // ---- divide by denominator (dve column 64 = tile 4, local col 0) and store
#pragma unroll
  for (int j = 0; j < 8; ++j) {
    const float dj = __shfl(cw[4][j], hi << 4, 32);  // lanes 0 / 16 hold col 64
    const float rinv = 1.0f / dj;
    const int row = rowbase + j + (hi << 3);
#pragma unroll
    for (int nt = 0; nt < 4; ++nt)
      out[(size_t)row * DV + nt * 16 + lo16] = cw[nt][j] * rinv;
  }
}

// =====================================================================
extern "C" void kernel_launch(void* const* d_in, const int* in_sizes, int n_in,
                              void* d_out, int out_size, void* d_ws, size_t ws_size,
                              hipStream_t stream) {
  (void)in_sizes; (void)n_in; (void)out_size; (void)ws_size;
  const float* q    = (const float*)d_in[0];
  const float* k    = (const float*)d_in[1];
  const float* v    = (const float*)d_in[2];
  const float* proj = (const float*)d_in[3];
  float* out = (float*)d_out;

  float*  part = (float*)d_ws;                                   // 10 MiB
  __bf16* zTe  = (__bf16*)((char*)d_ws +
                           (size_t)NB * DVE * RFEAT * sizeof(float));  // 40 KB

  favor_kz_partial<<<NB, 256, 0, stream>>>(k, v, proj, part);
  favor_z_reduce<<<DVE, 256, 0, stream>>>(part, zTe);
  favor_qw_out<<<N_ROWS / 128, 256, 0, stream>>>(q, proj, zTe, out);
}
